// HeteroGNN_32581621907694
// MI455X (gfx1250) — matrix operand order repaired and checked by reading, not compiled, not run on hardware
//
#include <hip/hip_runtime.h>
#include <hip/hip_bf16.h>
#include <math.h>

typedef __attribute__((ext_vector_type(16))) _Float16 v16h;
typedef __attribute__((ext_vector_type(8)))  float    v8f;

#define HID     64
#define HC      256        // H*C
#define NHEADS  4
#define NIP     100000
#define NPORT   10000
#define NPROTO  64
#define NGRAPH  64
#define NEGSLP  0.2f
#define SM_EPS  1e-16f

// ---------------- ordered-uint encoding for float atomic max ----------------
__device__ __forceinline__ unsigned f2ord(float f) {
  unsigned u = __float_as_uint(f);
  return (u & 0x80000000u) ? ~u : (u | 0x80000000u);
}
__device__ __forceinline__ float ord2f(unsigned u) {
  return __uint_as_float((u & 0x80000000u) ? (u & 0x7FFFFFFFu) : ~u);
}

// ---------------- generic fill ----------------
__global__ void k_fill(unsigned* __restrict__ p, unsigned v, size_t n) {
  size_t i = (size_t)blockIdx.x * blockDim.x + threadIdx.x;
  if (i < n) p[i] = v;
}

// ---------------- input linear: Y[N,64] = X[N,K] @ W[K,64] + b ----------------
__global__ void k_lin(const float* __restrict__ X, const float* __restrict__ W,
                      const float* __restrict__ b, float* __restrict__ Y,
                      int N, int K) {
  int tid = blockIdx.x * blockDim.x + threadIdx.x;
  if (tid >= N * HID) return;
  int n = tid >> 6, j = tid & 63;
  const float* xr = X + (size_t)n * K;
  float s = b[j];
  for (int k = 0; k < K; ++k) s = fmaf(xr[k], W[k * HID + j], s);
  Y[tid] = s;
}

// ---- pre-swizzle W[64,256] f32 -> f16 B fragments in WMMA register layout ----
// Wf layout: [t(16 N-tiles)][lane(32)][s(2 k-steps)] of v16h (16 contiguous halves).
// Lane fragment: n = t*16 + (lane&15); K base = s*32 + (lane<16 ? 0 : 8);
// halves (2v,2v+1) = K = base+16*(v>>2)+2*(v&3), +1.
__global__ void k_swz(const float* __restrict__ W, v16h* __restrict__ Wf) {
  int tid = blockIdx.x * blockDim.x + threadIdx.x;   // 1024 = 16*32*2
  if (tid >= 16 * 32 * 2) return;
  int s    = tid & 1;
  int lane = (tid >> 1) & 31;
  int t    = tid >> 6;
  int kb = s * 32 + ((lane < 16) ? 0 : 8);
  int n  = t * 16 + (lane & 15);
  v16h v;
#pragma unroll
  for (int vv = 0; vv < 8; ++vv) {
    int k0 = kb + ((vv >> 2) << 4) + ((vv & 3) << 1);
    v[2 * vv]     = (_Float16)W[(size_t)k0 * HC + n];
    v[2 * vv + 1] = (_Float16)W[(size_t)(k0 + 1) * HC + n];
  }
  Wf[tid] = v;   // ((t*32 + lane)*2 + s), 32B contiguous per lane fragment
}

// ---------------- WMMA GEMM: Y[M,256] = X[M,64] @ W[64,256] + bias ----------------
// blockDim = 128 (4 waves). Each wave computes one 16-row strip across all 256 cols.
// K=64 -> two 16x16x32 f16 WMMAs per 16x16 tile, f32 accumulate. B fragments are
// pre-swizzled f16 (k_swz) so each k-step is one contiguous 32B vector load.
__global__ void k_gemm_wmma(const float* __restrict__ X, const v16h* __restrict__ Wf,
                            const float* __restrict__ bias, float* __restrict__ Y,
                            int tileRows) {
  const int wave = threadIdx.x >> 5;
  const int lane = threadIdx.x & 31;
  const int tm = blockIdx.x * 4 + wave;
  if (tm >= tileRows) return;                    // wave-uniform; EXEC stays all-1

  // A fragment: 16x32 f16. lane -> M = lane&15 ; K base 0/8 by lane half.
  const int mrow = tm * 16 + (lane & 15);
  const int kbA  = (lane < 16) ? 0 : 8;
  const float* xrow = X + (size_t)mrow * HID;
  v16h a0, a1;
#pragma unroll
  for (int v = 0; v < 8; ++v) {
    int k0 = ((v >> 2) << 4) + kbA + ((v & 3) << 1);   // {0..7,16..23} or {8..15,24..31}
    a0[2 * v]     = (_Float16)xrow[k0];
    a0[2 * v + 1] = (_Float16)xrow[k0 + 1];
    a1[2 * v]     = (_Float16)xrow[32 + k0];
    a1[2 * v + 1] = (_Float16)xrow[32 + k0 + 1];
  }

  const int nl = lane & 15;
  const int mbase = tm * 16 + ((lane < 16) ? 0 : 8);
  const v16h* wlane = Wf + lane * 2;

#pragma unroll 1
  for (int t = 0; t < 16; ++t) {
    const int ncol = t * 16 + nl;
    if (t < 15) __builtin_prefetch(&wlane[(t + 1) * 64], 0, 0);
    v16h b0 = wlane[t * 64 + 0];
    v16h b1 = wlane[t * 64 + 1];
    v8f c = {};
    c = __builtin_amdgcn_wmma_f32_16x16x32_f16(false, a0, false, b0, (short)0, c, false, false);
    c = __builtin_amdgcn_wmma_f32_16x16x32_f16(false, a1, false, b1, (short)0, c, false, false);
    const float bv = bias[ncol];
#pragma unroll
    for (int i = 0; i < 8; ++i)
      Y[(size_t)(mbase + i) * HC + ncol] = c[i] + bv;
  }
}

// ---------------- edge logit + segment max ----------------
__global__ void k_edge_logit(const float* __restrict__ xl, const float* __restrict__ xr,
                             const int* __restrict__ src, const int* __restrict__ dst,
                             const float* __restrict__ att, float* __restrict__ logit,
                             unsigned* __restrict__ mmax, int E) {
  int tid = blockIdx.x * blockDim.x + threadIdx.x;
  int e = tid >> 2, h = tid & 3;
  if (e >= E) return;
  const float* pl = xl + (size_t)src[e] * HC + h * HID;
  const float* pr = xr + (size_t)dst[e] * HC + h * HID;
  const float* pa = att + h * HID;
  float s = 0.f;
#pragma unroll 4
  for (int c = 0; c < HID; ++c) {
    float t = pl[c] + pr[c];
    t = (t > 0.f) ? t : NEGSLP * t;
    s = fmaf(t, pa[c], s);
  }
  logit[(size_t)e * 4 + h] = s;
  atomicMax(&mmax[(size_t)dst[e] * 4 + h], f2ord(s));
}

// ---------------- w = exp(logit - m[dst]); denom[dst] += w ----------------
__global__ void k_edge_exp(float* __restrict__ logit, const int* __restrict__ dst,
                           const unsigned* __restrict__ mmax, float* __restrict__ denom, int E) {
  int tid = blockIdx.x * blockDim.x + threadIdx.x;
  int e = tid >> 2, h = tid & 3;
  if (e >= E) return;
  size_t i = (size_t)e * 4 + h;
  size_t di = (size_t)dst[e] * 4 + h;
  float w = expf(logit[i] - ord2f(mmax[di]));
  logit[i] = w;
  atomicAdd(&denom[di], w);
}

// ---------------- alpha = w / (denom[dst] + eps) ----------------
__global__ void k_edge_alpha(float* __restrict__ logit, const int* __restrict__ dst,
                             const float* __restrict__ denom, int E) {
  int tid = blockIdx.x * blockDim.x + threadIdx.x;
  int e = tid >> 2, h = tid & 3;
  if (e >= E) return;
  size_t i = (size_t)e * 4 + h;
  logit[i] = logit[i] / (denom[(size_t)dst[e] * 4 + h] + SM_EPS);
}

// ---------------- acc[dst,c] += 0.25 * sum_h alpha[e,h] * xl[src, h*64+c] ----------------
__global__ void k_edge_scatter(const float* __restrict__ xl, const float* __restrict__ alpha,
                               const int* __restrict__ src, const int* __restrict__ dst,
                               float* __restrict__ acc, int E) {
  int tid = blockIdx.x * blockDim.x + threadIdx.x;
  int e = tid >> 6, c = tid & 63;
  if (e >= E) return;
  const float* pl = xl + (size_t)src[e] * HC + c;
  const float* pa = alpha + (size_t)e * 4;
  float msg = 0.25f * (pa[0] * pl[0] + pa[1] * pl[64] + pa[2] * pl[128] + pa[3] * pl[192]);
  atomicAdd(&acc[(size_t)dst[e] * HID + c], msg);
}

// ---------------- out = elu(scale * (acc + b1 + b2?)) ----------------
__global__ void k_finalize(const float* __restrict__ acc, const float* __restrict__ b1,
                           const float* __restrict__ b2, float scale,
                           float* __restrict__ out, int N) {
  int tid = blockIdx.x * blockDim.x + threadIdx.x;
  if (tid >= N * HID) return;
  int c = tid & 63;
  float v = acc[tid] + b1[c] + (b2 ? b2[c] : 0.f);
  v *= scale;
  out[tid] = (v > 0.f) ? v : (expf(v) - 1.f);
}

// ---------------- global mean pool (sum + count) ----------------
__global__ void k_pool(const float* __restrict__ hip_, const int* __restrict__ batch,
                       float* __restrict__ pooled, float* __restrict__ cnt) {
  int tid = blockIdx.x * blockDim.x + threadIdx.x;
  if (tid >= NIP * HID) return;
  int n = tid >> 6, c = tid & 63;
  int g = batch[n];
  atomicAdd(&pooled[g * HID + c], hip_[tid]);
  if (c == 0) atomicAdd(&cnt[g], 1.0f);
}

// ---------------- classifier: 64 graphs, 64->32 relu -> 2 ----------------
__global__ void k_cls(const float* __restrict__ pooled, const float* __restrict__ cnt,
                      const float* __restrict__ W1, const float* __restrict__ b1,
                      const float* __restrict__ W2, const float* __restrict__ b2,
                      float* __restrict__ out) {
  int g = threadIdx.x;
  if (g >= NGRAPH) return;
  float inv = 1.0f / fmaxf(cnt[g], 1.0f);
  float x[HID];
  for (int k = 0; k < HID; ++k) x[k] = pooled[g * HID + k] * inv;
  float hbuf[32];
  for (int j = 0; j < 32; ++j) {
    float s = b1[j];
    for (int k = 0; k < HID; ++k) s = fmaf(x[k], W1[k * 32 + j], s);
    hbuf[j] = fmaxf(s, 0.f);
  }
  for (int m = 0; m < 2; ++m) {
    float s = b2[m];
    for (int j = 0; j < 32; ++j) s = fmaf(hbuf[j], W2[j * 2 + m], s);
    out[g * 2 + m] = s;
  }
}

// ---------------- one GATv2 relation ----------------
static void run_gat(const float* hsrc, int Ns, const float* hdst, int Nd,
                    const int* src, const int* dst, int E,
                    const float* Wl, const float* bl, const float* Wr, const float* br,
                    const float* att,
                    v16h* wf, float* xl, float* xr, float* logit, unsigned* mmax,
                    float* denom, float* acc, hipStream_t stream) {
  k_swz<<<4, 256, 0, stream>>>(Wl, wf);
  k_gemm_wmma<<<(Ns / 16 + 3) / 4, 128, 0, stream>>>(hsrc, wf, bl, xl, Ns / 16);
  k_swz<<<4, 256, 0, stream>>>(Wr, wf);
  k_gemm_wmma<<<(Nd / 16 + 3) / 4, 128, 0, stream>>>(hdst, wf, br, xr, Nd / 16);
  size_t nm = (size_t)Nd * 4;
  k_fill<<<(unsigned)((nm + 255) / 256), 256, 0, stream>>>(mmax, 0x007FFFFFu, nm);      // f2ord(-inf)
  k_fill<<<(unsigned)((nm + 255) / 256), 256, 0, stream>>>((unsigned*)denom, 0u, nm);
  size_t t4 = (size_t)E * 4;
  k_edge_logit<<<(unsigned)((t4 + 255) / 256), 256, 0, stream>>>(xl, xr, src, dst, att, logit, mmax, E);
  k_edge_exp  <<<(unsigned)((t4 + 255) / 256), 256, 0, stream>>>(logit, dst, mmax, denom, E);
  k_edge_alpha<<<(unsigned)((t4 + 255) / 256), 256, 0, stream>>>(logit, dst, denom, E);
  size_t t64 = (size_t)E * 64;
  k_edge_scatter<<<(unsigned)((t64 + 255) / 256), 256, 0, stream>>>(xl, logit, src, dst, acc, E);
}

extern "C" void kernel_launch(void* const* d_in, const int* in_sizes, int n_in,
                              void* d_out, int out_size, void* d_ws, size_t ws_size,
                              hipStream_t stream) {
  (void)in_sizes; (void)n_in; (void)out_size; (void)ws_size;
  // ---- inputs (setup_inputs() flattened order) ----
  const float* x_ip    = (const float*)d_in[0];
  const float* x_port  = (const float*)d_in[1];
  const float* x_proto = (const float*)d_in[2];
  const int* ei[10];
  for (int i = 0; i < 10; ++i) ei[i] = (const int*)d_in[3 + i];
  // ei: [0..1]=ipip, [2..3]=uses, [4..5]=runs, [6..7]=usedby, [8..9]=served (src,dst)
  const int* batch_ip = (const int*)d_in[13];
  const float* Wip = (const float*)d_in[14]; const float* bip = (const float*)d_in[15];
  const float* Wpo = (const float*)d_in[16]; const float* bpo = (const float*)d_in[17];
  const float* Wpr = (const float*)d_in[18]; const float* bpr = (const float*)d_in[19];
  // conv params: 20 + 30*layer + 6*rel + {Wl,bl,Wr,br,att,bias}; rel: ipip,uses,runs,usedby,served
  auto cp = [&](int layer, int rel, int which) -> const float* {
    return (const float*)d_in[20 + 30 * layer + 6 * rel + which];
  };
  const float* cW1 = (const float*)d_in[80]; const float* cb1 = (const float*)d_in[81];
  const float* cW2 = (const float*)d_in[82]; const float* cb2 = (const float*)d_in[83];
  float* out = (float*)d_out;

  // ---- workspace carve-up (256B aligned) ----
  char* ws = (char*)d_ws;
  size_t off = 0;
  auto carve = [&](size_t bytes) -> void* {
    void* p = ws + off;
    off += (bytes + 255) & ~(size_t)255;
    return p;
  };
  float* h_ip    = (float*)carve((size_t)NIP * HID * 4);
  float* h_ip2   = (float*)carve((size_t)NIP * HID * 4);
  float* h_port  = (float*)carve((size_t)NPORT * HID * 4);
  float* h_port2 = (float*)carve((size_t)NPORT * HID * 4);
  float* h_pro   = (float*)carve((size_t)NPROTO * HID * 4);
  float* h_pro2  = (float*)carve((size_t)NPROTO * HID * 4);
  float* xl      = (float*)carve((size_t)NIP * HC * 4);
  float* xr      = (float*)carve((size_t)NIP * HC * 4);
  float* logit   = (float*)carve((size_t)400000 * 4 * 4);
  unsigned* mmax = (unsigned*)carve((size_t)NIP * 4 * 4);
  float* denom   = (float*)carve((size_t)NIP * 4 * 4);
  float* acc_ip  = (float*)carve((size_t)NIP * HID * 4);
  float* acc_po  = (float*)carve((size_t)NPORT * HID * 4);
  float* acc_pr  = (float*)carve((size_t)NPROTO * HID * 4);
  float* pooled  = (float*)carve((size_t)NGRAPH * HID * 4);
  float* cnt     = (float*)carve((size_t)NGRAPH * 4);
  v16h*  wf      = (v16h*)carve((size_t)16 * 32 * 2 * sizeof(v16h));  // 32KB swizzled W

  // ---- input linears ----
  k_lin<<<(NIP * HID + 255) / 256, 256, 0, stream>>>(x_ip, Wip, bip, h_ip, NIP, 12);
  k_lin<<<(NPORT * HID + 255) / 256, 256, 0, stream>>>(x_port, Wpo, bpo, h_port, NPORT, 6);
  k_lin<<<(NPROTO * HID + 255) / 256, 256, 0, stream>>>(x_proto, Wpr, bpr, h_pro, NPROTO, 5);

  float* cip = h_ip, *nip = h_ip2;
  float* cpo = h_port, *npo = h_port2;
  float* cpr = h_pro, *npr = h_pro2;

  for (int L = 0; L < 2; ++L) {
    k_fill<<<(NIP * HID + 255) / 256, 256, 0, stream>>>((unsigned*)acc_ip, 0u, (size_t)NIP * HID);
    k_fill<<<(NPORT * HID + 255) / 256, 256, 0, stream>>>((unsigned*)acc_po, 0u, (size_t)NPORT * HID);
    k_fill<<<(NPROTO * HID + 255) / 256, 256, 0, stream>>>((unsigned*)acc_pr, 0u, (size_t)NPROTO * HID);

    // -> ip: ipip (rel 0) + usedby (rel 3)
    run_gat(cip, NIP, cip, NIP, ei[0], ei[1], 400000,
            cp(L,0,0), cp(L,0,1), cp(L,0,2), cp(L,0,3), cp(L,0,4),
            wf, xl, xr, logit, mmax, denom, acc_ip, stream);
    run_gat(cpo, NPORT, cip, NIP, ei[6], ei[7], 200000,
            cp(L,3,0), cp(L,3,1), cp(L,3,2), cp(L,3,3), cp(L,3,4),
            wf, xl, xr, logit, mmax, denom, acc_ip, stream);
    // -> port: uses (rel 1) + served (rel 4)
    run_gat(cip, NIP, cpo, NPORT, ei[2], ei[3], 200000,
            cp(L,1,0), cp(L,1,1), cp(L,1,2), cp(L,1,3), cp(L,1,4),
            wf, xl, xr, logit, mmax, denom, acc_po, stream);
    run_gat(cpr, NPROTO, cpo, NPORT, ei[8], ei[9], 50000,
            cp(L,4,0), cp(L,4,1), cp(L,4,2), cp(L,4,3), cp(L,4,4),
            wf, xl, xr, logit, mmax, denom, acc_po, stream);
    // -> proto: runs (rel 2)
    run_gat(cpo, NPORT, cpr, NPROTO, ei[4], ei[5], 50000,
            cp(L,2,0), cp(L,2,1), cp(L,2,2), cp(L,2,3), cp(L,2,4),
            wf, xl, xr, logit, mmax, denom, acc_pr, stream);

    k_finalize<<<(NIP * HID + 255) / 256, 256, 0, stream>>>(acc_ip, cp(L,0,5), cp(L,3,5), 0.5f, nip, NIP);
    k_finalize<<<(NPORT * HID + 255) / 256, 256, 0, stream>>>(acc_po, cp(L,1,5), cp(L,4,5), 0.5f, npo, NPORT);
    k_finalize<<<(NPROTO * HID + 255) / 256, 256, 0, stream>>>(acc_pr, cp(L,2,5), (const float*)nullptr, 1.0f, npr, NPROTO);

    float* t;
    t = cip; cip = nip; nip = t;
    t = cpo; cpo = npo; npo = t;
    t = cpr; cpr = npr; npr = t;
  }

  // ---- global mean pool over ip nodes + classifier ----
  k_fill<<<(NGRAPH * HID + 255) / 256, 256, 0, stream>>>((unsigned*)pooled, 0u, (size_t)NGRAPH * HID);
  k_fill<<<1, 256, 0, stream>>>((unsigned*)cnt, 0u, (size_t)NGRAPH);
  k_pool<<<(NIP * HID + 255) / 256, 256, 0, stream>>>(cip, batch_ip, pooled, cnt);
  k_cls<<<1, 64, 0, stream>>>(pooled, cnt, cW1, cb1, cW2, cb2, out);
}
